// Multi_Linear_16587163697284
// MI455X (gfx1250) — matrix operand order repaired and checked by reading, not compiled, not run on hardware
//
#include <hip/hip_runtime.h>
#include <hip/hip_bf16.h>

typedef __attribute__((ext_vector_type(16))) __bf16 v16bf;
typedef __attribute__((ext_vector_type(8)))  float  v8f;

#define BB   16
#define CC   32
#define NN   207
#define CIN  64
#define COUT 64
#define NPAIR (CC*NN)   // 6624

__global__ __launch_bounds__(256) void multi_linear_wmma(
    const float* __restrict__ x,    // [B,C,N,CIN]
    const float* __restrict__ rw,   // [C,N,CIN,COUT,2]
    const float* __restrict__ rb,   // [1,C,N,1,1]
    const float* __restrict__ mw,   // [COUT,CIN]
    const float* __restrict__ mb,   // [COUT]
    const float* __restrict__ gw,   // [COUT,CIN]
    const float* __restrict__ gb,   // [COUT]
    const float* __restrict__ ow,   // [2]
    const float* __restrict__ ob,   // [1]
    float* __restrict__ out)        // [B,C,N,COUT]
{
    const int lane = threadIdx.x & 31;
    const int wid  = threadIdx.x >> 5;
    const int pair = blockIdx.x * 8 + wid;           // (c,n) index, wave-uniform
    if (pair >= NPAIR) return;                       // uniform -> EXEC stays all-1s

    const int c    = pair / NN;
    const int n    = pair - c * NN;
    const int half = lane >> 4;                      // lane 0-15 vs 16-31
    const int l16  = lane & 15;

    const float bn    = 0.99999500003749968f;        // 1/sqrt(1+1e-5)
    const float ow0   = ow[0], ow1 = ow[1];
    const float ubias = rb[pair] * (ow0 + ow1) + ob[0];

    // ---- A fragments: row m = l16 of X; h0 = x, h1 = x*x*bn (bf16) ----
    const float* xrow = x + ((size_t)(l16 * CC + c) * NN + n) * CIN;
    v16bf ax[2], ax2[2];
#pragma unroll
    for (int ks = 0; ks < 2; ++ks) {
#pragma unroll
        for (int v = 0; v < 8; ++v) {
            // A 16-bit layout: K-pair base = (v>>2)*16 + half*8 + (v&3)*2
            const int k = ks * 32 + ((v >> 2) << 4) + (half << 3) + ((v & 3) << 1);
            const float x0 = xrow[k];
            const float x1 = xrow[k + 1];
            ax [ks][2 * v]     = (__bf16)x0;
            ax [ks][2 * v + 1] = (__bf16)x1;
            ax2[ks][2 * v]     = (__bf16)(x0 * x0 * bn);
            ax2[ks][2 * v + 1] = (__bf16)(x1 * x1 * bn);
        }
    }

    const float* wbase = rw + (size_t)pair * CIN * COUT * 2;   // contiguous 32KB tile

#pragma unroll 1
    for (int nt = 0; nt < 4; ++nt) {
        v8f u0 = {}, u1 = {}, s = {}, g = {};
        const int o = nt * 16 + l16;                 // this lane's output column

#pragma unroll
        for (int ks = 0; ks < 2; ++ks) {
            v16bf bw0, bw1, bm, bg;
#pragma unroll
            for (int v = 0; v < 8; ++v) {
                // B 16-bit layout: K-pair base = (v>>2)*16 + half*8 + (v&3)*2
                const int k = ks * 32 + ((v >> 2) << 4) + (half << 3) + ((v & 3) << 1);
                // grouped weights: element (k,o,d) at (k*64+o)*2+d -> b64 grabs d0,d1
                const float* p0 = wbase + ((size_t)k * COUT + o) * 2;
                const float* p1 = p0 + COUT * 2;     // row k+1
                bw0[2 * v]     = (__bf16)p0[0];
                bw1[2 * v]     = (__bf16)p0[1];
                bw0[2 * v + 1] = (__bf16)p1[0];
                bw1[2 * v + 1] = (__bf16)p1[1];
                // shared/gate: B[k,o] = W[o,k] (k,k+1 contiguous -> b64)
                const float* pm = mw + (size_t)o * CIN + k;
                const float* pg = gw + (size_t)o * CIN + k;
                bm[2 * v]     = (__bf16)pm[0];
                bm[2 * v + 1] = (__bf16)pm[1];
                bg[2 * v]     = (__bf16)pg[0];
                bg[2 * v + 1] = (__bf16)pg[1];
            }
            u0 = __builtin_amdgcn_wmma_f32_16x16x32_bf16(false, ax [ks], false, bw0, (short)0, u0, false, false);
            u1 = __builtin_amdgcn_wmma_f32_16x16x32_bf16(false, ax2[ks], false, bw1, (short)0, u1, false, false);
            s  = __builtin_amdgcn_wmma_f32_16x16x32_bf16(false, ax [ks], false, bm,  (short)0, s,  false, false);
            g  = __builtin_amdgcn_wmma_f32_16x16x32_bf16(false, ax [ks], false, bg,  (short)0, g,  false, false);
        }

        // ---- epilogue: D layout -> lane holds (m = half*8 + r, o) ----
        const float mbo = mb[o];
        const float gbo = gb[o];
#pragma unroll
        for (int r = 0; r < 8; ++r) {
            const int m = half * 8 + r;
            const size_t base = ((size_t)(m * CC + c) * NN + n);
            const float xres = x[base * CIN + o];                 // residual (CIN==COUT)
            const float uu   = (u0[r] * ow0 + u1[r] * ow1 + ubias) * 0.125f; // SCALE/4
            const float sh   = s[r] + mbo;
            const float gate = 1.0f / (1.0f + __expf(-(g[r] + gbo)));
            out[base * COUT + o] = gate * uu + (1.0f - gate) * sh + xres;
        }
    }
}

extern "C" void kernel_launch(void* const* d_in, const int* in_sizes, int n_in,
                              void* d_out, int out_size, void* d_ws, size_t ws_size,
                              hipStream_t stream) {
    const float* x  = (const float*)d_in[0];
    const float* rw = (const float*)d_in[1];
    const float* rb = (const float*)d_in[2];
    const float* mw = (const float*)d_in[3];
    const float* mb = (const float*)d_in[4];
    const float* gw = (const float*)d_in[5];
    const float* gb = (const float*)d_in[6];
    const float* ow = (const float*)d_in[7];
    const float* ob = (const float*)d_in[8];
    float* out = (float*)d_out;

    // 6624 (c,n) pairs, one wave32 each; 8 waves per 256-thread block
    dim3 grid((NPAIR + 7) / 8);   // 828 blocks * 8 waves = 6624
    dim3 block(256);
    multi_linear_wmma<<<grid, block, 0, stream>>>(x, rw, rb, mw, mb, gw, gb, ow, ob, out);
}
